// MessagePassing_22986664968611
// MI455X (gfx1250) — matrix operand order repaired and checked by reading, not compiled, not run on hardware
//
#include <hip/hip_runtime.h>
#include <hip/hip_bf16.h>

typedef _Float16 v8h  __attribute__((ext_vector_type(8)));
typedef _Float16 v16h __attribute__((ext_vector_type(16)));
typedef float    v8f  __attribute__((ext_vector_type(8)));

// Problem constants (match reference)
static constexpr long long N_ = 50000;   // nodes
static constexpr long long E_ = 500000;  // edges
static constexpr long long B_ = 16;      // graphs
static constexpr int       F_ = 128;     // all output feature widths
static constexpr float     BN_EPS = 1e-5f;

// ---------------------------------------------------------------------------
// Weight conversion: f32 [K,128] row-major  ->  f16 transposed [128][K]
// so each WMMA B-fragment (fixed n, contiguous k) is one 32-byte load.
// ---------------------------------------------------------------------------
__global__ void convert_weights_kernel(const float* __restrict__ We,
                                       const float* __restrict__ Wn,
                                       const float* __restrict__ Wg,
                                       _Float16* __restrict__ WtE,
                                       _Float16* __restrict__ WtN,
                                       _Float16* __restrict__ WtG) {
  int idx = blockIdx.x * 256 + threadIdx.x;
  if (idx < 128 * 320) {
    int n = idx / 320, k = idx % 320;
    WtE[idx] = (_Float16)We[k * 128 + n];
  }
  if (idx < 128 * 384) {
    int n = idx / 384, k = idx % 384;
    WtN[idx] = (_Float16)Wn[k * 128 + n];
    WtG[idx] = (_Float16)Wg[k * 128 + n];
  }
}

// ---------------------------------------------------------------------------
// Degree / per-graph counts for the three scatter-means.
// ---------------------------------------------------------------------------
__global__ void counts_kernel(const long long* __restrict__ edge_index,
                              const long long* __restrict__ batch,
                              float* __restrict__ deg,    // [N] incoming-edge count
                              float* __restrict__ ecnt,   // [B] edges per graph
                              float* __restrict__ ncnt) { // [B] nodes per graph
  long long i = (long long)blockIdx.x * 256 + threadIdx.x;
  if (i < E_) {
    atomicAdd(&deg[edge_index[E_ + i]], 1.0f);            // dst
    atomicAdd(&ecnt[batch[edge_index[i]]], 1.0f);         // batch[src]
  }
  if (i < N_) atomicAdd(&ncnt[batch[i]], 1.0f);
}

// ---------------------------------------------------------------------------
// Fused gather + GEMM (WMMA f32<-f16 16x16x32) + bias + BN-stat reduction.
// MODE 0: edge update   A = concat(x[src], edge_attr, u[batch[src]])  K=320
// MODE 1: node update   A = concat(x, msg_sum/deg, u[batch])          K=384
// Block = 256 threads = 8 waves; one 16-row tile per block; wave w owns
// output features [16w, 16w+16). Writes pre-BN activations to out_pre.
// ---------------------------------------------------------------------------
template <int MODE, int K>
__global__ __launch_bounds__(256)
void gemm_tile_kernel(const float* __restrict__ x,
                      const float* __restrict__ edge_attr,
                      const float* __restrict__ u,
                      const long long* __restrict__ edge_index,
                      const long long* __restrict__ batch,
                      const float* __restrict__ msg_sum,   // MODE 1
                      const float* __restrict__ deg,       // MODE 1
                      const _Float16* __restrict__ Wt,     // [128][K] f16
                      const float* __restrict__ bias,      // [128]
                      float* __restrict__ out_pre,         // [rows][128]
                      float* __restrict__ statSum,         // [128]
                      float* __restrict__ statSS) {        // [128]
  constexpr int KT = K / 32;
  constexpr int KP = K + 8;                    // keep rows 16B-aligned, depad banks
  __shared__ alignas(16) _Float16 Ain[16][KP];
  __shared__ long long rowSrc[16];
  __shared__ long long rowBat[16];
  __shared__ float rowScale[16];
  __shared__ float sSum[128];
  __shared__ float sSS[128];

  const int t = threadIdx.x;
  const long long row0 = (long long)blockIdx.x * 16;

  if (t < 16) {
    long long r = row0 + t;
    if (MODE == 0) {
      long long s = edge_index[r];             // src node of edge r
      rowSrc[t] = s;
      rowBat[t] = batch[s];
    } else {
      rowBat[t] = batch[r];
      rowScale[t] = 1.0f / fmaxf(deg[r], 1.0f);
    }
  }
  __syncthreads();

  // Stage concatenated A tile into LDS as f16.
  for (int idx = t; idx < 16 * K; idx += 256) {
    int r = idx / K, c = idx % K;
    long long row = row0 + r;
    float v;
    if (MODE == 0) {
      if (c < 128)      v = x[rowSrc[r] * 128 + c];
      else if (c < 192) v = edge_attr[row * 64 + (c - 128)];
      else              v = u[rowBat[r] * 128 + (c - 192)];
    } else {
      if (c < 128)      v = x[row * 128 + c];
      else if (c < 256) v = msg_sum[row * 128 + (c - 128)] * rowScale[r];
      else              v = u[rowBat[r] * 128 + (c - 256)];
    }
    Ain[r][c] = (_Float16)v;
  }
  __syncthreads();

  const int wave = t >> 5;
  const int lane = t & 31;
  const int hi   = lane >> 4;         // lane group (0: lanes 0-15, 1: 16-31)
  const int n    = wave * 16 + (lane & 15);

  v8f acc = {0.f, 0.f, 0.f, 0.f, 0.f, 0.f, 0.f, 0.f};
  const _Float16* wcol = Wt + (long long)n * K;

#pragma unroll
  for (int kt = 0; kt < KT; ++kt) {
    // A fragment: contiguous k = kt*32 + hi*16 + [0..15] (matches B).
    int ka = kt * 32 + (hi << 4);
    v8h alo = *(const v8h*)&Ain[lane & 15][ka];
    v8h ahi = *(const v8h*)&Ain[lane & 15][ka + 8];
    v16h a = __builtin_shufflevector(alo, ahi, 0, 1, 2, 3, 4, 5, 6, 7,
                                     8, 9, 10, 11, 12, 13, 14, 15);
    // B fragment: fixed column n, contiguous k = kt*32 + hi*16 + [0..15].
    v16h b = *(const v16h*)(wcol + kt * 32 + (hi << 4));
    acc = __builtin_amdgcn_wmma_f32_16x16x32_f16(false, a, false, b,
                                                 (short)0, acc, false, false);
  }

  // Epilogue: + bias, store pre-BN, accumulate per-feature sum / sum^2.
  const float bv = bias[n];
  float s = 0.f, ss = 0.f;
#pragma unroll
  for (int j = 0; j < 8; ++j) {
    float h = acc[j] + bv;
    long long m = row0 + j + hi * 8;           // C/D layout: M = j + 8*hi
    out_pre[m * 128 + n] = h;
    s += h;
    ss += h * h;
  }
  s  += __shfl_xor(s, 16, 32);
  ss += __shfl_xor(ss, 16, 32);
  if (hi == 0) { sSum[n] = s; sSS[n] = ss; }
  __syncthreads();
  if (t < 128) {
    atomicAdd(&statSum[t], sSum[t]);
    atomicAdd(&statSS[t], sSS[t]);
  }
}

// ---------------------------------------------------------------------------
// Fold batch stats into per-feature scale/shift: y = h*scale + shift.
// ---------------------------------------------------------------------------
__global__ void finalize_stats_kernel(const float* __restrict__ sum,
                                      const float* __restrict__ ss,
                                      const float* __restrict__ gamma,
                                      const float* __restrict__ beta,
                                      float* __restrict__ scale,
                                      float* __restrict__ shift,
                                      float count) {
  int i = threadIdx.x;                          // 128 threads
  float mean = sum[i] / count;
  float var  = ss[i] / count - mean * mean;
  float sc   = rsqrtf(var + BN_EPS) * gamma[i];
  scale[i] = sc;
  shift[i] = beta[i] - mean * sc;
}

// ---------------------------------------------------------------------------
// Edge BN+ReLU in place + fused scatter-sums into msg_sum[dst], eg_sum[gb].
// ---------------------------------------------------------------------------
__global__ void edge_apply_kernel(float* __restrict__ e_io,
                                  const float* __restrict__ scale,
                                  const float* __restrict__ shift,
                                  const long long* __restrict__ edge_index,
                                  const long long* __restrict__ batch,
                                  float* __restrict__ msg_sum,
                                  float* __restrict__ eg_sum) {
  long long idx = (long long)blockIdx.x * 256 + threadIdx.x;
  long long e = idx >> 7;
  int f = (int)(idx & 127);
  float v = fmaxf(e_io[idx] * scale[f] + shift[f], 0.0f);
  e_io[idx] = v;
  long long dst = edge_index[E_ + e];
  atomicAdd(&msg_sum[dst * 128 + f], v);
  long long gb = batch[edge_index[e]];
  atomicAdd(&eg_sum[gb * 128 + f], v);
}

// ---------------------------------------------------------------------------
// Node BN+ReLU in place + fused scatter-sum into xg_sum[batch].
// ---------------------------------------------------------------------------
__global__ void node_apply_kernel(float* __restrict__ x_io,
                                  const float* __restrict__ scale,
                                  const float* __restrict__ shift,
                                  const long long* __restrict__ batch,
                                  float* __restrict__ xg_sum) {
  long long idx = (long long)blockIdx.x * 256 + threadIdx.x;
  long long nidx = idx >> 7;
  int f = (int)(idx & 127);
  float v = fmaxf(x_io[idx] * scale[f] + shift[f], 0.0f);
  x_io[idx] = v;
  atomicAdd(&xg_sum[batch[nidx] * 128 + f], v);
}

// ---------------------------------------------------------------------------
// Global update: exactly one 16-row WMMA tile (B=16 graphs). One block does
// staging, GEMM (K=384), in-register BN over 16 rows, ReLU, store.
// ---------------------------------------------------------------------------
__global__ __launch_bounds__(256)
void global_kernel(const float* __restrict__ eg_sum, const float* __restrict__ ecnt,
                   const float* __restrict__ xg_sum, const float* __restrict__ ncnt,
                   const float* __restrict__ u, const _Float16* __restrict__ WtG,
                   const float* __restrict__ bias, const float* __restrict__ gamma,
                   const float* __restrict__ beta, float* __restrict__ u_new) {
  constexpr int K = 384;
  __shared__ alignas(16) _Float16 Ain[16][K + 8];
  const int t = threadIdx.x;
  for (int idx = t; idx < 16 * K; idx += 256) {
    int r = idx / K, c = idx % K;
    float v;
    if (c < 128)      v = eg_sum[r * 128 + c] / fmaxf(ecnt[r], 1.0f);
    else if (c < 256) v = xg_sum[r * 128 + (c - 128)] / fmaxf(ncnt[r], 1.0f);
    else              v = u[r * 128 + (c - 256)];
    Ain[r][c] = (_Float16)v;
  }
  __syncthreads();

  const int wave = t >> 5;
  const int lane = t & 31;
  const int hi   = lane >> 4;
  const int n    = wave * 16 + (lane & 15);

  v8f acc = {0.f, 0.f, 0.f, 0.f, 0.f, 0.f, 0.f, 0.f};
  const _Float16* wcol = WtG + (long long)n * K;
#pragma unroll
  for (int kt = 0; kt < K / 32; ++kt) {
    int ka = kt * 32 + (hi << 4);
    v8h alo = *(const v8h*)&Ain[lane & 15][ka];
    v8h ahi = *(const v8h*)&Ain[lane & 15][ka + 8];
    v16h a = __builtin_shufflevector(alo, ahi, 0, 1, 2, 3, 4, 5, 6, 7,
                                     8, 9, 10, 11, 12, 13, 14, 15);
    v16h b = *(const v16h*)(wcol + kt * 32 + (hi << 4));
    acc = __builtin_amdgcn_wmma_f32_16x16x32_f16(false, a, false, b,
                                                 (short)0, acc, false, false);
  }

  const float bv = bias[n];
  float h[8], s = 0.f, ss = 0.f;
#pragma unroll
  for (int j = 0; j < 8; ++j) {
    h[j] = acc[j] + bv;
    s += h[j];
    ss += h[j] * h[j];
  }
  s  += __shfl_xor(s, 16, 32);                  // combine the two row halves
  ss += __shfl_xor(ss, 16, 32);
  float mean = s * (1.0f / 16.0f);
  float var  = ss * (1.0f / 16.0f) - mean * mean;
  float sc   = rsqrtf(var + BN_EPS) * gamma[n];
  float sh   = beta[n] - mean * sc;
#pragma unroll
  for (int j = 0; j < 8; ++j)
    u_new[(long long)(j + hi * 8) * 128 + n] = fmaxf(h[j] * sc + sh, 0.0f);
}

// ---------------------------------------------------------------------------
// Pass-through outputs (edge_index, batch) as floats.
// ---------------------------------------------------------------------------
__global__ void meta_kernel(const long long* __restrict__ edge_index,
                            const long long* __restrict__ batch,
                            float* __restrict__ out_ei,
                            float* __restrict__ out_batch) {
  long long i = (long long)blockIdx.x * 256 + threadIdx.x;
  if (i < 2 * E_) out_ei[i] = (float)edge_index[i];
  if (i < N_) out_batch[i] = (float)batch[i];
}

// ---------------------------------------------------------------------------
extern "C" void kernel_launch(void* const* d_in, const int* in_sizes, int n_in,
                              void* d_out, int out_size, void* d_ws, size_t ws_size,
                              hipStream_t stream) {
  (void)in_sizes; (void)n_in; (void)out_size; (void)ws_size;

  const float*     x        = (const float*)d_in[0];
  const float*     edgeA    = (const float*)d_in[1];
  const float*     u        = (const float*)d_in[2];
  const float*     W_e      = (const float*)d_in[3];
  const float*     b_e      = (const float*)d_in[4];
  const float*     g_e      = (const float*)d_in[5];
  const float*     beta_e   = (const float*)d_in[6];
  const float*     W_n      = (const float*)d_in[7];
  const float*     b_n      = (const float*)d_in[8];
  const float*     g_n      = (const float*)d_in[9];
  const float*     beta_n   = (const float*)d_in[10];
  const float*     W_g      = (const float*)d_in[11];
  const float*     b_g      = (const float*)d_in[12];
  const float*     g_g      = (const float*)d_in[13];
  const float*     beta_g   = (const float*)d_in[14];
  const long long* edge_idx = (const long long*)d_in[15];
  const long long* batch    = (const long long*)d_in[16];

  // d_out layout (flat, return order): x_new, e, edge_index, u_new, batch
  float* out   = (float*)d_out;
  float* out_x = out;                   // N*128 (holds n_pre, then x_new in place)
  float* out_e = out_x + N_ * F_;       // E*128 (holds e_pre, then e in place)
  float* out_i = out_e + E_ * F_;       // 2*E
  float* out_u = out_i + 2 * E_;        // B*128
  float* out_b = out_u + B_ * F_;       // N

  // Workspace: zeroed accumulators first, then f16 weights.
  float* ws      = (float*)d_ws;
  float* msg_sum = ws;                  // [N,128]
  float* deg     = msg_sum + N_ * F_;   // [N]
  float* eg_sum  = deg + N_;            // [B,128]
  float* ecnt    = eg_sum + B_ * F_;    // [B]
  float* xg_sum  = ecnt + B_;           // [B,128]
  float* ncnt    = xg_sum + B_ * F_;    // [B]
  float* sumE    = ncnt + B_;           // [128]
  float* ssE     = sumE + 128;
  float* scE     = ssE + 128;
  float* shE     = scE + 128;
  float* sumN    = shE + 128;
  float* ssN     = sumN + 128;
  float* scN     = ssN + 128;
  float* shN     = scN + 128;
  _Float16* WtE  = (_Float16*)(shN + 128);   // [128][320]
  _Float16* WtN  = WtE + 128 * 320;          // [128][384]
  _Float16* WtG  = WtN + 128 * 384;          // [128][384]

  size_t zbytes = (size_t)((char*)WtE - (char*)d_ws);
  hipMemsetAsync(d_ws, 0, zbytes, stream);

  convert_weights_kernel<<<(128 * 384 + 255) / 256, 256, 0, stream>>>(
      W_e, W_n, W_g, WtE, WtN, WtG);

  counts_kernel<<<(int)((E_ + 255) / 256), 256, 0, stream>>>(
      edge_idx, batch, deg, ecnt, ncnt);

  // Edge update: e_pre = concat(x[src], edge_attr, u[batch[src]]) @ W_e + b_e
  gemm_tile_kernel<0, 320><<<(int)(E_ / 16), 256, 0, stream>>>(
      x, edgeA, u, edge_idx, batch, nullptr, nullptr, WtE, b_e, out_e, sumE, ssE);
  finalize_stats_kernel<<<1, 128, 0, stream>>>(sumE, ssE, g_e, beta_e, scE, shE,
                                               (float)E_);
  edge_apply_kernel<<<(int)(E_ * F_ / 256), 256, 0, stream>>>(
      out_e, scE, shE, edge_idx, batch, msg_sum, eg_sum);

  // Node update: n_pre = concat(x, msg_sum/deg, u[batch]) @ W_n + b_n
  gemm_tile_kernel<1, 384><<<(int)(N_ / 16), 256, 0, stream>>>(
      x, edgeA, u, edge_idx, batch, msg_sum, deg, WtN, b_n, out_x, sumN, ssN);
  finalize_stats_kernel<<<1, 128, 0, stream>>>(sumN, ssN, g_n, beta_n, scN, shN,
                                               (float)N_);
  node_apply_kernel<<<(int)(N_ * F_ / 256), 256, 0, stream>>>(
      out_x, scN, shN, batch, xg_sum);

  // Global update: one 16-row WMMA tile, fused BN+ReLU.
  global_kernel<<<1, 256, 0, stream>>>(eg_sum, ecnt, xg_sum, ncnt, u, WtG,
                                       b_g, g_g, beta_g, out_u);

  // Pass-through outputs.
  meta_kernel<<<(int)((2 * E_ + 255) / 256), 256, 0, stream>>>(edge_idx, batch,
                                                               out_i, out_b);
}
